// BioMIP_6305011990650
// MI455X (gfx1250) — compile-verified, matrix operand order checked by reading, not run on hardware
//
#include <hip/hip_runtime.h>

// ---------------------------------------------------------------------------
// CDNA5 (gfx1250) implementation of BioMIP: AttentiveFP x2 + 2-layer RGCN.
// Dense GEMMs run on v_wmma_f32_16x16x32_f16 (f16 A/B staged in LDS, f32 acc).
// ---------------------------------------------------------------------------

typedef __attribute__((ext_vector_type(16))) _Float16 v16h;
typedef __attribute__((ext_vector_type(8)))  float    v8f;

#define D_EMB 128
#define N_DRUG 2048
#define N_TGT  1024

union FragU { unsigned u[8]; v16h v; };
union HQuad { _Float16 h[4]; ushort4 u; };

// ------------------------- WMMA GEMM (with row gather) ---------------------
// C[M,N] = act( (gather(A) @ B) [+ C if accum] [+ bias] )
// A: [M?,K] row-major (rows picked via gidx if non-null), B: [K,N] row-major.
// Block = 256 thr = 8 waves; tile = 128x64 per block, K-step 32.
// Each wave owns one 16-row strip and four 16x16 N-quarters: A fragment is
// loaded once and reused for 4 WMMAs per K-step (XDL-dense inner loop).
#define TILE_M 128
#define TILE_N 64
#define TILE_K 32
#define LPAD   36   // halves per LDS row (72B stride: 8B-aligned, conflict-free)

__global__ __launch_bounds__(256)
void gemm_wmma_kernel(const float* __restrict__ A, int lda, const int* __restrict__ gidx,
                      const float* __restrict__ B, int ldb,
                      const float* __restrict__ bias,
                      float* __restrict__ C, int ldc,
                      int M, int N, int K, int accum, int act)
{
    __shared__ _Float16 As[TILE_M][LPAD];   // ~9.2 KB
    __shared__ _Float16 Bs[TILE_N][LPAD];   // ~4.6 KB (transposed: [n][k])

    const int t    = threadIdx.x;
    const int wave = t >> 5;
    const int lane = t & 31;
    const int m0   = blockIdx.y * TILE_M;
    const int n0   = blockIdx.x * TILE_N;

    const bool fast = (m0 + TILE_M <= M) && (n0 + TILE_N <= N) && ((K & (TILE_K - 1)) == 0);

    // fast path: each thread stages 4 float4 of A (rows arow+32i) and 2 float4 of B.
    const int arow = t >> 3;            // 0..31
    const int acol = (t & 7) * 4;       // 0,4,..,28
    const float* aptr[4];
    if (fast) {
        #pragma unroll
        for (int i = 0; i < 4; ++i) {
            int gm = m0 + arow + 32 * i;
            int row = gidx ? gidx[gm] : gm;
            aptr[i] = A + (long)row * lda + acol;   // gather hoisted out of K loop
        }
    }

    v8f acc[4] = {{}, {}, {}, {}};
    const int nk = (K + TILE_K - 1) / TILE_K;
    for (int kt = 0; kt < nk; ++kt) {
        const int k0 = kt * TILE_K;
        if (fast) {
            // A tile: 128x32, unpredicated b128 loads, packed 8B LDS stores
            #pragma unroll
            for (int i = 0; i < 4; ++i) {
                float4 v = *(const float4*)(aptr[i] + k0);
                HQuad q;
                q.h[0] = (_Float16)v.x; q.h[1] = (_Float16)v.y;
                q.h[2] = (_Float16)v.z; q.h[3] = (_Float16)v.w;
                *(ushort4*)&As[arow + 32 * i][acol] = q.u;
            }
            // B tile: 32x64, b128 loads, transposed scalar stores
            #pragma unroll
            for (int i = 0; i < 2; ++i) {
                int lin = t + 256 * i;
                int kk = lin >> 4, n4 = (lin & 15) * 4;
                float4 v = *(const float4*)(B + (long)(k0 + kk) * ldb + n0 + n4);
                Bs[n4 + 0][kk] = (_Float16)v.x;
                Bs[n4 + 1][kk] = (_Float16)v.y;
                Bs[n4 + 2][kk] = (_Float16)v.z;
                Bs[n4 + 3][kk] = (_Float16)v.w;
            }
        } else {
            // ragged tile (feature GEMMs: K=39/21/10/11): branchless clamped loads
            #pragma unroll
            for (int i = 0; i < 16; ++i) {
                int lin = t + 256 * i;
                int r = lin >> 5, c = lin & 31;
                int gm = m0 + r;
                bool mv = gm < M, kv = (k0 + c) < K;
                int gmc = mv ? gm : 0;
                int row = gidx ? gidx[gmc] : gmc;
                float v = A[(long)row * lda + (kv ? k0 + c : 0)];
                As[r][c] = (_Float16)((mv && kv) ? v : 0.0f);
            }
            #pragma unroll
            for (int i = 0; i < 8; ++i) {
                int lin = t + 256 * i;
                int kk = lin >> 6, nn = lin & 63;
                bool kv = (k0 + kk) < K, nv = (n0 + nn) < N;
                float v = B[(long)(kv ? k0 + kk : 0) * ldb + (nv ? n0 + nn : 0)];
                Bs[nn][kk] = (_Float16)((kv && nv) ? v : 0.0f);
            }
        }
        if (kt + 1 < nk)  // hint next B slab into cache (global_prefetch_b8)
            __builtin_prefetch(B + (size_t)(k0 + TILE_K) * ldb + n0, 0, 1);
        __syncthreads();

        // pack fragments per ISA 7.12.2 16-bit layouts (wave32)
        const int g = lane >> 4, mr = lane & 15;
        FragU af, bf[4];
        #pragma unroll
        for (int j = 0; j < 8; ++j) {
            int k2 = ((j < 4) ? (2 * j) : (2 * j + 8)) + 8 * g;  // K pair base
            af.u[j] = *(const unsigned*)&As[wave * 16 + mr][k2];
            #pragma unroll
            for (int q = 0; q < 4; ++q)
                bf[q].u[j] = *(const unsigned*)&Bs[16 * q + mr][k2];
        }
        #pragma unroll
        for (int q = 0; q < 4; ++q)
            acc[q] = __builtin_amdgcn_wmma_f32_16x16x32_f16(false, af.v, false, bf[q].v,
                                                            (short)0, acc[q], false, false);
        __syncthreads();
    }

    // store: lane holds C[m0+wave*16 + r + 8*(lane/16)][n0 + 16*q + lane&15]
    const int g = lane >> 4;
    #pragma unroll
    for (int q = 0; q < 4; ++q) {
        int n = n0 + 16 * q + (lane & 15);
        #pragma unroll
        for (int r = 0; r < 8; ++r) {
            int m = m0 + wave * 16 + r + 8 * g;
            if (m < M && n < N) {
                float v = acc[q][r];
                if (accum) v += C[(long)m * ldc + n];
                if (bias)  v += bias[n];
                if (act == 1)      v = v > 0.0f ? v : 0.2f * v;   // leaky relu
                else if (act == 2) v = v > 0.0f ? v : 0.0f;       // relu
                C[(long)m * ldc + n] = v;
            }
        }
    }
}

// ------------------------- per-edge attention logit ------------------------
// out[e] = lrelu( dot(X[xi], att[0:128]) + dot(Y[yi], att[128:256]) )
__global__ __launch_bounds__(256)
void edge_att_kernel(const float* __restrict__ X, const int* __restrict__ xidx,
                     const float* __restrict__ Y, const int* __restrict__ yidx,
                     const float* __restrict__ att, float* __restrict__ out, long E)
{
    long e = (long)blockIdx.x * 8 + (threadIdx.x >> 5);
    int lane = threadIdx.x & 31;
    if (e >= E) return;
    long xr = xidx ? xidx[e] : e;
    long yr = yidx ? yidx[e] : e;
    float4 xv = ((const float4*)(X + xr * D_EMB))[lane];
    float4 yv = ((const float4*)(Y + yr * D_EMB))[lane];
    float4 a0 = ((const float4*)att)[lane];
    float4 a1 = ((const float4*)(att + D_EMB))[lane];
    float s = xv.x * a0.x + xv.y * a0.y + xv.z * a0.z + xv.w * a0.w
            + yv.x * a1.x + yv.y * a1.y + yv.z * a1.z + yv.w * a1.w;
    #pragma unroll
    for (int off = 16; off > 0; off >>= 1) s += __shfl_xor(s, off, 32);
    if (lane == 0) out[e] = s > 0.0f ? s : 0.2f * s;
}

// ------------------------- segment softmax pieces --------------------------
__device__ __forceinline__ unsigned encf(float f) {
    unsigned u = __float_as_uint(f);
    return (u & 0x80000000u) ? ~u : (u | 0x80000000u);     // monotone encoding
}
__device__ __forceinline__ float decf(unsigned u) {
    u = (u & 0x80000000u) ? (u & 0x7FFFFFFFu) : ~u;
    return __uint_as_float(u);
}

__global__ void fill_u32_kernel(unsigned* p, unsigned v, long n) {
    long i = (long)blockIdx.x * 256 + threadIdx.x;
    if (i < n) p[i] = v;
}

__global__ void segmax_kernel(const float* lg, const int* seg, unsigned* menc, long E) {
    long i = (long)blockIdx.x * 256 + threadIdx.x;
    if (i < E) atomicMax(&menc[seg[i]], encf(lg[i]));
}

__global__ void segexp_kernel(const float* lg, const int* seg, const unsigned* menc,
                              float* ssum, float* ex, long E) {
    long i = (long)blockIdx.x * 256 + threadIdx.x;
    if (i >= E) return;
    int s = seg[i];
    float v = __expf(lg[i] - decf(menc[s]));
    ex[i] = v;
    atomicAdd(&ssum[s], v);
}

// ctx[seg[e]] += (ex[e]/(ssum[seg]+1e-9)) * P[e]   (over D=128)
__global__ void wsegsum_kernel(const float* __restrict__ ex, const float* __restrict__ ssum,
                               const int* __restrict__ seg, const float* __restrict__ P,
                               float* __restrict__ ctx, long E) {
    long i = (long)blockIdx.x * 256 + threadIdx.x;
    if (i >= E * D_EMB) return;
    long e = i >> 7; int d = i & 127;
    int sg = seg[e];
    float a = ex[e] / (ssum[sg] + 1e-9f);
    atomicAdd(ctx + (long)sg * D_EMB + d, a * P[e * D_EMB + d]);
}

// out[seg[n]] += x[n]   (over D=128)
__global__ void segsum_kernel(const float* __restrict__ x, const int* __restrict__ seg,
                              float* __restrict__ out, long N) {
    long i = (long)blockIdx.x * 256 + threadIdx.x;
    if (i >= N * D_EMB) return;
    long n = i >> 7; int d = i & 127;
    atomicAdd(out + (long)seg[n] * D_EMB + d, x[i]);
}

__global__ void count_kernel(const int* seg, float* cnt, long E) {
    long i = (long)blockIdx.x * 256 + threadIdx.x;
    if (i < E) atomicAdd(&cnt[seg[i]], 1.0f);
}

__global__ void combine_mean_kernel(float* out, const float* agg, const float* cnt, long N128) {
    long i = (long)blockIdx.x * 256 + threadIdx.x;
    if (i >= N128) return;
    out[i] += agg[i] / fmaxf(cnt[i >> 7], 1.0f);
}

__global__ void act_kernel(float* x, long n, int act) {
    long i = (long)blockIdx.x * 256 + threadIdx.x;
    if (i >= n) return;
    float v = x[i];
    if (act == 2)      v = v > 0.0f ? v : 0.0f;         // relu
    else if (act == 3) v = v > 0.0f ? v : expm1f(v);    // elu
    x[i] = v;
}

// GRU: h' = (1-z)*n + z*h ; gx holds [z|r|n] pre-acts incl. bias, gh likewise.
__global__ void gru_kernel(const float* __restrict__ gx, const float* __restrict__ gh,
                           const float* __restrict__ h, float* __restrict__ out, long N) {
    long i = (long)blockIdx.x * 256 + threadIdx.x;
    if (i >= N * D_EMB) return;
    long n = i >> 7; int d = i & 127;
    const float* gxr = gx + n * 3 * D_EMB;
    const float* ghr = gh + n * 3 * D_EMB;
    float z  = 1.0f / (1.0f + __expf(-(gxr[d] + ghr[d])));
    float r  = 1.0f / (1.0f + __expf(-(gxr[D_EMB + d] + ghr[D_EMB + d])));
    float nn = tanhf(gxr[2 * D_EMB + d] + r * ghr[2 * D_EMB + d]);
    out[i] = (1.0f - z) * nn + z * h[i];
}

// ---------------------------- host-side helpers ----------------------------
static void gemm(hipStream_t st, const float* A, int lda, const int* gidx,
                 const float* B, int ldb, const float* bias,
                 float* C, int ldc, int M, int N, int K, int accum, int act) {
    dim3 grid((N + TILE_N - 1) / TILE_N, (M + TILE_M - 1) / TILE_M);
    gemm_wmma_kernel<<<grid, 256, 0, st>>>(A, lda, gidx, B, ldb, bias, C, ldc,
                                           M, N, K, accum, act);
}
static void fillu(hipStream_t st, unsigned* p, unsigned v, long n) {
    fill_u32_kernel<<<(unsigned)((n + 255) / 256), 256, 0, st>>>(p, v, n);
}
static void segsoftmax(hipStream_t st, const float* lg, float* ex, const int* seg,
                       long E, long nseg, unsigned* menc, float* ssum) {
    fillu(st, menc, 0x007FFFFFu, nseg);          // enc(-inf)
    fillu(st, (unsigned*)ssum, 0u, nseg);
    segmax_kernel<<<(unsigned)((E + 255) / 256), 256, 0, st>>>(lg, seg, menc, E);
    segexp_kernel<<<(unsigned)((E + 255) / 256), 256, 0, st>>>(lg, seg, menc, ssum, ex, E);
}

struct GruP { const float *Wh, *Wx, *b; };
struct AfpP {
    const float *W_edge, *W_node, *att0, *att_r;
    GruP gru0, gru_r;
    const float* lat[2]; GruP lgru[2]; const float* lproj[2];
    const float *proj0, *proj_r;
};

static void run_afp(hipStream_t st, const float* nfeat, int fn, const float* efeat, int fe,
                    const int* src, const int* dst, const int* gid,
                    long Nn, long En, int G, const AfpP& p, float* g_out,
                    float* hv, float* A1, float* A2, float* ctx,
                    float* lg, float* ex, unsigned* menc, float* ssum)
{
    const unsigned BN = (unsigned)((Nn * D_EMB + 255) / 256);
    const unsigned BE = (unsigned)((En * D_EMB + 255) / 256);
    // hv = lrelu(x @ W_node)
    gemm(st, nfeat, fn, nullptr, p.W_node, D_EMB, nullptr, hv, D_EMB, (int)Nn, D_EMB, fn, 0, 1);
    // he = lrelu(x[src]@We[:fn] + e@We[fn:])   -> A1
    gemm(st, nfeat, fn, src, p.W_edge, D_EMB, nullptr, A1, D_EMB, (int)En, D_EMB, fn, 0, 0);
    gemm(st, efeat, fe, nullptr, p.W_edge + (long)fn * D_EMB, D_EMB, nullptr,
         A1, D_EMB, (int)En, D_EMB, fe, 1, 1);
    // step0 attention + context + GRU
    edge_att_kernel<<<(unsigned)((En + 7) / 8), 256, 0, st>>>(hv, dst, A1, nullptr, p.att0, lg, En);
    segsoftmax(st, lg, ex, dst, En, Nn, menc, ssum);
    gemm(st, A1, D_EMB, nullptr, p.proj0, D_EMB, nullptr, A2, D_EMB, (int)En, D_EMB, D_EMB, 0, 0);
    fillu(st, (unsigned*)ctx, 0u, Nn * D_EMB);
    wsegsum_kernel<<<BE, 256, 0, st>>>(ex, ssum, dst, A2, ctx, En);
    act_kernel<<<BN, 256, 0, st>>>(ctx, Nn * D_EMB, 3);
    gemm(st, ctx, D_EMB, nullptr, p.gru0.Wx, 3 * D_EMB, p.gru0.b, A2, 3 * D_EMB, (int)Nn, 3 * D_EMB, D_EMB, 0, 0);
    gemm(st, hv,  D_EMB, nullptr, p.gru0.Wh, 3 * D_EMB, nullptr,  A1, 3 * D_EMB, (int)Nn, 3 * D_EMB, D_EMB, 0, 0);
    gru_kernel<<<BN, 256, 0, st>>>(A2, A1, hv, hv, Nn);
    // GNN timesteps
    for (int l = 0; l < 2; ++l) {
        edge_att_kernel<<<(unsigned)((En + 7) / 8), 256, 0, st>>>(hv, dst, hv, src, p.lat[l], lg, En);
        segsoftmax(st, lg, ex, dst, En, Nn, menc, ssum);
        gemm(st, hv, D_EMB, src, p.lproj[l], D_EMB, nullptr, A2, D_EMB, (int)En, D_EMB, D_EMB, 0, 0);
        fillu(st, (unsigned*)ctx, 0u, Nn * D_EMB);
        wsegsum_kernel<<<BE, 256, 0, st>>>(ex, ssum, dst, A2, ctx, En);
        act_kernel<<<BN, 256, 0, st>>>(ctx, Nn * D_EMB, 3);
        gemm(st, ctx, D_EMB, nullptr, p.lgru[l].Wx, 3 * D_EMB, p.lgru[l].b, A2, 3 * D_EMB, (int)Nn, 3 * D_EMB, D_EMB, 0, 0);
        gemm(st, hv,  D_EMB, nullptr, p.lgru[l].Wh, 3 * D_EMB, nullptr,     A1, 3 * D_EMB, (int)Nn, 3 * D_EMB, D_EMB, 0, 0);
        gru_kernel<<<BN, 256, 0, st>>>(A2, A1, hv, hv, Nn);
    }
    // attentive readout: g = segsum(h), then 2 GRU steps over super-node
    const unsigned BG = (unsigned)(((long)G * D_EMB + 255) / 256);
    fillu(st, (unsigned*)g_out, 0u, (long)G * D_EMB);
    segsum_kernel<<<BN, 256, 0, st>>>(hv, gid, g_out, Nn);
    for (int t = 0; t < 2; ++t) {
        edge_att_kernel<<<(unsigned)((Nn + 7) / 8), 256, 0, st>>>(g_out, gid, hv, nullptr, p.att_r, lg, Nn);
        segsoftmax(st, lg, ex, gid, Nn, G, menc, ssum);
        gemm(st, hv, D_EMB, nullptr, p.proj_r, D_EMB, nullptr, A2, D_EMB, (int)Nn, D_EMB, D_EMB, 0, 0);
        fillu(st, (unsigned*)ctx, 0u, (long)G * D_EMB);
        wsegsum_kernel<<<BN, 256, 0, st>>>(ex, ssum, gid, A2, ctx, Nn);
        act_kernel<<<BG, 256, 0, st>>>(ctx, (long)G * D_EMB, 3);
        gemm(st, ctx,   D_EMB, nullptr, p.gru_r.Wx, 3 * D_EMB, p.gru_r.b, A2, 3 * D_EMB, G, 3 * D_EMB, D_EMB, 0, 0);
        gemm(st, g_out, D_EMB, nullptr, p.gru_r.Wh, 3 * D_EMB, nullptr,   A1, 3 * D_EMB, G, 3 * D_EMB, D_EMB, 0, 0);
        gru_kernel<<<BG, 256, 0, st>>>(A2, A1, g_out, g_out, G);
    }
}

struct RgcnLayerP { const float* rel[4]; const float* self_d; const float* self_t; };
struct RelEdges  { const int* src; const int* dst; int E; int stype; int dtype; };

static void rgcn_layer(hipStream_t st, const float* hd, const float* ht,
                       const RgcnLayerP& L, const RelEdges* re,
                       float* outd, float* outt,
                       float* mbuf, float* agg, float* cnt, int relu)
{
    gemm(st, hd, D_EMB, nullptr, L.self_d, D_EMB, nullptr, outd, D_EMB, N_DRUG, D_EMB, D_EMB, 0, 0);
    gemm(st, ht, D_EMB, nullptr, L.self_t, D_EMB, nullptr, outt, D_EMB, N_TGT,  D_EMB, D_EMB, 0, 0);
    for (int r = 0; r < 4; ++r) {
        const float* hs = re[r].stype ? ht : hd;
        int nd = re[r].dtype ? N_TGT : N_DRUG;
        float* outp = re[r].dtype ? outt : outd;
        long E = re[r].E;
        gemm(st, hs, D_EMB, re[r].src, L.rel[r], D_EMB, nullptr, mbuf, D_EMB, (int)E, D_EMB, D_EMB, 0, 0);
        fillu(st, (unsigned*)agg, 0u, (long)nd * D_EMB);
        fillu(st, (unsigned*)cnt, 0u, nd);
        segsum_kernel<<<(unsigned)((E * D_EMB + 255) / 256), 256, 0, st>>>(mbuf, re[r].dst, agg, E);
        count_kernel<<<(unsigned)((E + 255) / 256), 256, 0, st>>>(re[r].dst, cnt, E);
        combine_mean_kernel<<<(unsigned)(((long)nd * D_EMB + 255) / 256), 256, 0, st>>>(outp, agg, cnt, (long)nd * D_EMB);
    }
    if (relu) {
        act_kernel<<<(unsigned)(((long)N_DRUG * D_EMB + 255) / 256), 256, 0, st>>>(outd, (long)N_DRUG * D_EMB, 2);
        act_kernel<<<(unsigned)(((long)N_TGT  * D_EMB + 255) / 256), 256, 0, st>>>(outt, (long)N_TGT  * D_EMB, 2);
    }
}

// --------------------------------- entry -----------------------------------
extern "C" void kernel_launch(void* const* d_in, const int* in_sizes, int n_in,
                              void* d_out, int out_size, void* d_ws, size_t ws_size,
                              hipStream_t stream)
{
    (void)n_in; (void)out_size; (void)ws_size;
    // graph tensors
    const float* s_nf = (const float*)d_in[0];
    const float* s_ef = (const float*)d_in[1];
    const int *s_src = (const int*)d_in[2], *s_dst = (const int*)d_in[3], *s_gid = (const int*)d_in[4];
    const float* t_nf = (const float*)d_in[5];
    const float* t_ef = (const float*)d_in[6];
    const int *t_src = (const int*)d_in[7], *t_dst = (const int*)d_in[8], *t_gid = (const int*)d_in[9];
    long NS = in_sizes[0] / 39, ES = in_sizes[2];
    long NT = in_sizes[5] / 21, ET = in_sizes[7];

    RelEdges posE[4], negE[4];
    for (int j = 0; j < 4; ++j) {
        static const int styp[4] = {0, 0, 1, 1}, dtyp[4] = {0, 1, 0, 1}; // dd,dt,td,tt
        posE[j] = { (const int*)d_in[10 + 2 * j], (const int*)d_in[11 + 2 * j],
                    in_sizes[10 + 2 * j], styp[j], dtyp[j] };
        negE[j] = { (const int*)d_in[18 + 2 * j], (const int*)d_in[19 + 2 * j],
                    in_sizes[18 + 2 * j], styp[j], dtyp[j] };
    }

    // params: jax pytree order (dict keys sorted recursively)
    int idx = 26;
    auto nf = [&]() { return (const float*)d_in[idx++]; };
    auto parse_afp = [&](AfpP& p) {
        p.W_edge = nf(); p.W_node = nf(); p.att0 = nf(); p.att_r = nf();
        p.gru0.Wh = nf(); p.gru0.Wx = nf(); p.gru0.b = nf();
        p.gru_r.Wh = nf(); p.gru_r.Wx = nf(); p.gru_r.b = nf();
        for (int l = 0; l < 2; ++l) {
            p.lat[l] = nf(); p.lgru[l].Wh = nf(); p.lgru[l].Wx = nf(); p.lgru[l].b = nf();
            p.lproj[l] = nf();
        }
        p.proj0 = nf(); p.proj_r = nf();
    };
    AfpP sp, mp; parse_afp(sp); parse_afp(mp);
    RgcnLayerP L1, L2;
    for (int l = 0; l < 2; ++l) {
        RgcnLayerP& L = l ? L2 : L1;
        for (int r = 0; r < 4; ++r) L.rel[r] = nf();
        L.self_d = nf(); L.self_t = nf();
    }

    // workspace bump allocator
    char* wp = (char*)d_ws;
    auto alloc = [&](size_t bytes) { void* r = (void*)wp; wp += (bytes + 255) & ~(size_t)255; return r; };
    float* drug_intra = (float*)alloc((size_t)N_DRUG * D_EMB * 4);
    float* tgt_intra  = (float*)alloc((size_t)N_TGT  * D_EMB * 4);
    char* mark = wp;

    // ---- AttentiveFP on small-molecule graph ----
    {
        long big = (ES * D_EMB > NS * 3 * D_EMB) ? ES * D_EMB : NS * 3 * D_EMB;
        float* hv  = (float*)alloc((size_t)NS * D_EMB * 4);
        float* A1  = (float*)alloc((size_t)big * 4);
        float* A2  = (float*)alloc((size_t)big * 4);
        float* ctx = (float*)alloc((size_t)NS * D_EMB * 4);
        float* lg  = (float*)alloc((size_t)ES * 4);
        float* ex  = (float*)alloc((size_t)ES * 4);
        unsigned* menc = (unsigned*)alloc((size_t)NS * 4);
        float* ssum = (float*)alloc((size_t)NS * 4);
        run_afp(stream, s_nf, 39, s_ef, 10, s_src, s_dst, s_gid, NS, ES, N_DRUG, sp,
                drug_intra, hv, A1, A2, ctx, lg, ex, menc, ssum);
    }
    wp = mark;
    // ---- AttentiveFP on target (protein) graph ----
    {
        long big = (ET * D_EMB > NT * 3 * D_EMB) ? ET * D_EMB : NT * 3 * D_EMB;
        float* hv  = (float*)alloc((size_t)NT * D_EMB * 4);
        float* A1  = (float*)alloc((size_t)big * 4);
        float* A2  = (float*)alloc((size_t)big * 4);
        float* ctx = (float*)alloc((size_t)NT * D_EMB * 4);
        float* lg  = (float*)alloc((size_t)ET * 4);
        float* ex  = (float*)alloc((size_t)ET * 4);
        unsigned* menc = (unsigned*)alloc((size_t)NT * 4);
        float* ssum = (float*)alloc((size_t)NT * 4);
        run_afp(stream, t_nf, 21, t_ef, 11, t_src, t_dst, t_gid, NT, ET, N_TGT, mp,
                tgt_intra, hv, A1, A2, ctx, lg, ex, menc, ssum);
    }
    wp = mark;
    // ---- RGCN over pos / neg edge sets ----
    {
        int maxE = 0;
        for (int r = 0; r < 4; ++r) { if (posE[r].E > maxE) maxE = posE[r].E; if (negE[r].E > maxE) maxE = negE[r].E; }
        float* h1d  = (float*)alloc((size_t)N_DRUG * D_EMB * 4);
        float* h1t  = (float*)alloc((size_t)N_TGT  * D_EMB * 4);
        float* mbuf = (float*)alloc((size_t)maxE * D_EMB * 4);
        float* agg  = (float*)alloc((size_t)N_DRUG * D_EMB * 4);
        float* cnt  = (float*)alloc((size_t)N_DRUG * 4);
        float* out = (float*)d_out;
        float* pos_d = out;
        float* pos_t = pos_d + (size_t)N_DRUG * D_EMB;
        float* neg_d = pos_t + (size_t)N_TGT * D_EMB;
        float* neg_t = neg_d + (size_t)N_DRUG * D_EMB;
        rgcn_layer(stream, drug_intra, tgt_intra, L1, posE, h1d, h1t, mbuf, agg, cnt, 1);
        rgcn_layer(stream, h1d, h1t,            L2, posE, pos_d, pos_t, mbuf, agg, cnt, 0);
        rgcn_layer(stream, drug_intra, tgt_intra, L1, negE, h1d, h1t, mbuf, agg, cnt, 1);
        rgcn_layer(stream, h1d, h1t,            L2, negE, neg_d, neg_t, mbuf, agg, cnt, 0);
    }
}
